// SelectiveTransformerLayer_69432441307315
// MI455X (gfx1250) — matrix operand order repaired and checked by reading, not compiled, not run on hardware
//
#include <hip/hip_runtime.h>
#include <hip/hip_bf16.h>
#include <stdint.h>
#include <stddef.h>

// ---------------------------------------------------------------------------
// MI455X (gfx1250) implementation. Compute-bound layer (~0.84 TFLOP vs ~0.4 GB
// traffic => ~2000 FLOP/B >> roofline knee at 23.3 TB/s), so all GEMMs and the
// attention core run on v_wmma_f32_16x16x32_bf16 (bf16 in, f32 accumulate).
// ---------------------------------------------------------------------------

typedef __bf16 bf16_t;
typedef __attribute__((ext_vector_type(16))) __bf16        v16bf;
typedef __attribute__((ext_vector_type(8)))  float         v8f;
typedef __attribute__((ext_vector_type(4)))  unsigned int  u32x4;
typedef __attribute__((ext_vector_type(4)))  float         f32x4;

#define HIDDEN 2048
#define NH     32
#define NKV    4
#define HD     128
#define DFF    8192
#define SEQ    2048
#define BATCH  2
#define MROWS  (BATCH * SEQ)          // 4096
#define ATTN_SCALE 0.08838834764831845f  // 1/sqrt(128)

__device__ __forceinline__ bf16_t f2bf(float x) {
  unsigned u = __builtin_bit_cast(unsigned, x);
  u += 0x7FFFu + ((u >> 16) & 1u);               // round-to-nearest-even
  unsigned short h = (unsigned short)(u >> 16);
  return __builtin_bit_cast(bf16_t, h);
}

union Frag { v16bf v; u32x4 q[2]; };

// ---------------------------------------------------------------------------
// fp32 -> bf16 weight conversion
// ---------------------------------------------------------------------------
__global__ __launch_bounds__(256) void cvt_kernel(const float* __restrict__ in,
                                                  bf16_t* __restrict__ out, int n) {
  int i = (blockIdx.x * 256 + threadIdx.x) * 4;
  if (i < n) {
    f32x4 v = *(const f32x4*)(in + i);
    out[i + 0] = f2bf(v[0]); out[i + 1] = f2bf(v[1]);
    out[i + 2] = f2bf(v[2]); out[i + 3] = f2bf(v[3]);
  }
}

// ---------------------------------------------------------------------------
// RMSNorm over HIDDEN, fp32 in -> bf16 out. One block per row.
// ---------------------------------------------------------------------------
__global__ __launch_bounds__(256) void rmsnorm_kernel(const float* __restrict__ x,
                                                      const float* __restrict__ w,
                                                      bf16_t* __restrict__ out) {
  const int row = blockIdx.x;
  const int tid = threadIdx.x;
  const float* xr = x + (size_t)row * HIDDEN;
  float ss = 0.f;
  for (int c = tid; c < HIDDEN; c += 256) { float v = xr[c]; ss += v * v; }
#pragma unroll
  for (int m = 16; m >= 1; m >>= 1) ss += __shfl_xor(ss, m, 32);
  __shared__ float red[8];
  if ((tid & 31) == 0) red[tid >> 5] = ss;
  __syncthreads();
  float tot = 0.f;
#pragma unroll
  for (int i = 0; i < 8; ++i) tot += red[i];
  float sc = rsqrtf(tot * (1.0f / HIDDEN) + 1e-6f);
  for (int c = tid; c < HIDDEN; c += 256)
    out[(size_t)row * HIDDEN + c] = f2bf(xr[c] * sc * w[c]);
}

// ---------------------------------------------------------------------------
// WMMA GEMM:  C[M,N] = A[M,K](bf16) * W[N,K](bf16)^T
// block tile 128x64, 8 waves (4x2), 32x32 per wave, BK=64.
// MODE 0: f32 out | 1: bf16 out | 2: f32 out + residual | 3: silu(g)*u -> bf16
// Fragment layouts per CDNA5 ISA 7.12.2 (wave32).
// ---------------------------------------------------------------------------
template <int MODE>
__global__ __launch_bounds__(256) void gemm_kernel(
    const bf16_t* __restrict__ A, const bf16_t* __restrict__ W,
    const bf16_t* __restrict__ W2, const float* __restrict__ RES,
    float* __restrict__ Cf, bf16_t* __restrict__ Cb, int M, int N, int K) {
  __shared__ bf16_t As[128][72];
  __shared__ bf16_t Ws[64][72];
  __shared__ bf16_t Ws2[64][72];

  const int tid  = threadIdx.x;
  const int wave = tid >> 5, lane = tid & 31;
  const int ln15 = lane & 15, hi = lane >> 4;
  const int wm = wave >> 1, wn = wave & 1;
  const int m0 = blockIdx.y * 128;
  const int n0 = blockIdx.x * 64;

  v8f acc[2][2]  = {};
  v8f acc2[2][2] = {};

  for (int k0 = 0; k0 < K; k0 += 64) {
#pragma unroll
    for (int t = 0; t < 4; ++t) {
      int lin = tid + t * 256;
      int r = lin >> 3, c = (lin & 7) * 8;
      *(u32x4*)&As[r][c] = *(const u32x4*)&A[(size_t)(m0 + r) * K + k0 + c];
    }
#pragma unroll
    for (int t = 0; t < 2; ++t) {
      int lin = tid + t * 256;
      int r = lin >> 3, c = (lin & 7) * 8;
      *(u32x4*)&Ws[r][c] = *(const u32x4*)&W[(size_t)(n0 + r) * K + k0 + c];
      if (MODE == 3)
        *(u32x4*)&Ws2[r][c] = *(const u32x4*)&W2[(size_t)(n0 + r) * K + k0 + c];
    }
    __syncthreads();

#pragma unroll
    for (int ks = 0; ks < 64; ks += 32) {
      Frag a[2], bw[2], b2[2];
#pragma unroll
      for (int tm = 0; tm < 2; ++tm) {
        int row = wm * 32 + tm * 16 + ln15;
        a[tm].q[0] = *(const u32x4*)&As[row][ks + hi * 8];
        a[tm].q[1] = *(const u32x4*)&As[row][ks + 16 + hi * 8];
      }
#pragma unroll
      for (int tn = 0; tn < 2; ++tn) {
        int nr = wn * 32 + tn * 16 + ln15;
        bw[tn].q[0] = *(const u32x4*)&Ws[nr][ks + hi * 8];
        bw[tn].q[1] = *(const u32x4*)&Ws[nr][ks + 16 + hi * 8];
        if (MODE == 3) {
          b2[tn].q[0] = *(const u32x4*)&Ws2[nr][ks + hi * 8];
          b2[tn].q[1] = *(const u32x4*)&Ws2[nr][ks + 16 + hi * 8];
        }
      }
#pragma unroll
      for (int tm = 0; tm < 2; ++tm)
#pragma unroll
        for (int tn = 0; tn < 2; ++tn) {
          acc[tm][tn] = __builtin_amdgcn_wmma_f32_16x16x32_bf16(
              false, a[tm].v, false, bw[tn].v, (short)0, acc[tm][tn], false, false);
          if (MODE == 3)
            acc2[tm][tn] = __builtin_amdgcn_wmma_f32_16x16x32_bf16(
                false, a[tm].v, false, b2[tn].v, (short)0, acc2[tm][tn], false, false);
        }
    }
    __syncthreads();
  }

#pragma unroll
  for (int tm = 0; tm < 2; ++tm)
#pragma unroll
    for (int tn = 0; tn < 2; ++tn)
#pragma unroll
      for (int r = 0; r < 8; ++r) {
        int row = m0 + wm * 32 + tm * 16 + r + hi * 8;
        int col = n0 + wn * 32 + tn * 16 + ln15;
        size_t idx = (size_t)row * N + col;
        float v = acc[tm][tn][r];
        if (MODE == 0) {
          Cf[idx] = v;
        } else if (MODE == 1) {
          Cb[idx] = f2bf(v);
        } else if (MODE == 2) {
          Cf[idx] = v + RES[idx];
        } else {
          float g = v, u = acc2[tm][tn][r];
          float s = g / (1.0f + __expf(-g));
          Cb[idx] = f2bf(s * u);
        }
      }
}

// ---------------------------------------------------------------------------
// Fused per-head QK RMSNorm + mRoPE. One wave per (b, s, head) row of 128.
// Lane holds d = lane + 32j (j=0..3); rotate_half pairing d <-> d+64 is
// lane-local (j0<->j2, j1<->j3).
// ---------------------------------------------------------------------------
__device__ __forceinline__ float mrope_freq(int d) {
  int sd, off;
  if (d < 48)      { sd = 48; off = d; }
  else if (d < 88) { sd = 40; off = d - 48; }
  else             { sd = 40; off = d - 88; }
  int fi = off % (sd >> 1);
  return __expf(-(2.0f * fi / (float)sd) * 13.815510557964274f);  // ln(1e6)
}

__global__ __launch_bounds__(256) void qknorm_rope_kernel(
    const float* __restrict__ qf, const float* __restrict__ kf,
    const float* __restrict__ qw, const float* __restrict__ kw,
    bf16_t* __restrict__ qb, bf16_t* __restrict__ kb) {
  const int lane = threadIdx.x & 31;
  const int gid  = blockIdx.x * 8 + (threadIdx.x >> 5);
  const int NQ = BATCH * SEQ * NH;
  const int NT = NQ + BATCH * SEQ * NKV;
  if (gid >= NT) return;

  const float* src; const float* nw; bf16_t* dst; int s;
  if (gid < NQ) {
    int b = gid / (SEQ * NH); int rem = gid % (SEQ * NH);
    s = rem / NH; int h = rem % NH;
    src = qf + (size_t)(b * SEQ + s) * (NH * HD) + h * HD;
    dst = qb + (size_t)((b * NH + h) * SEQ + s) * HD;
    nw  = qw;
  } else {
    int t = gid - NQ;
    int b = t / (SEQ * NKV); int rem = t % (SEQ * NKV);
    s = rem / NKV; int h = rem % NKV;
    src = kf + (size_t)(b * SEQ + s) * (NKV * HD) + h * HD;
    dst = kb + (size_t)((b * NKV + h) * SEQ + s) * HD;
    nw  = kw;
  }

  float v[4];
#pragma unroll
  for (int j = 0; j < 4; ++j) v[j] = src[lane + 32 * j];
  float ss = v[0]*v[0] + v[1]*v[1] + v[2]*v[2] + v[3]*v[3];
#pragma unroll
  for (int m = 16; m >= 1; m >>= 1) ss += __shfl_xor(ss, m, 32);
  float scn = rsqrtf(ss * (1.0f / HD) + 1e-6f);

  float xn[4];
#pragma unroll
  for (int j = 0; j < 4; ++j) xn[j] = v[j] * scn * nw[lane + 32 * j];
  float rot[4] = { -xn[2], -xn[3], xn[0], xn[1] };
#pragma unroll
  for (int j = 0; j < 4; ++j) {
    int d = lane + 32 * j;
    float ang = (float)s * mrope_freq(d);
    float sn, cs;
    sincosf(ang, &sn, &cs);
    dst[d] = f2bf(xn[j] * cs + rot[j] * sn);
  }
}

// ---------------------------------------------------------------------------
// Flash attention (causal, GQA). Block: (qtile of 128 rows, head, batch).
// 8 waves, wave w owns 16 q rows. Key blocks of 64 staged in LDS; V stored
// transposed so PV B-fragments are pairs of contiguous-K 16B LDS loads. Both QK^T and
// PV run on v_wmma_f32_16x16x32_bf16; P is routed C-layout -> A-layout
// through per-wave LDS with an explicit s_wait_dscnt.
// ---------------------------------------------------------------------------
__global__ __launch_bounds__(256) void flash_kernel(
    const bf16_t* __restrict__ Q, const bf16_t* __restrict__ Kv,
    const bf16_t* __restrict__ V, bf16_t* __restrict__ O) {
  __shared__ bf16_t Ks[64][136];
  __shared__ bf16_t Vt[128][72];
  __shared__ bf16_t Ps[128][72];

  const int qt = blockIdx.x, h = blockIdx.y, b = blockIdx.z;
  const int kvh  = h >> 3;                 // N_REP = 8
  const int tid  = threadIdx.x;
  const int wave = tid >> 5, lane = tid & 31;
  const int ln15 = lane & 15, hi = lane >> 4;
  const int qbase = qt * 128 + wave * 16;

  // Q A-fragments: 4 chunks of K=32 over head dim
  Frag qfr[4];
  {
    const bf16_t* qp = Q + ((size_t)(b * NH + h) * SEQ + qbase + ln15) * HD;
#pragma unroll
    for (int c = 0; c < 4; ++c) {
      qfr[c].q[0] = *(const u32x4*)(qp + c * 32 + hi * 8);
      qfr[c].q[1] = *(const u32x4*)(qp + c * 32 + 16 + hi * 8);
    }
  }

  v8f oacc[8] = {};
  float m_run[8], l_run[8];
#pragma unroll
  for (int r = 0; r < 8; ++r) { m_run[r] = -1e30f; l_run[r] = 0.f; }

  const int nkb = (qt + 1) * 2;  // 64-key blocks up to causal frontier
  for (int kb = 0; kb < nkb; ++kb) {
#pragma unroll
    for (int t = 0; t < 4; ++t) {
      int lin = tid + t * 256;
      int r = lin >> 4, c = (lin & 15) * 8;
      *(u32x4*)&Ks[r][c] =
          *(const u32x4*)&Kv[((size_t)(b * NKV + kvh) * SEQ + kb * 64 + r) * HD + c];
    }
#pragma unroll
    for (int t = 0; t < 4; ++t) {
      int lin = tid + t * 256;
      int key = lin >> 4, c = (lin & 15) * 8;
      u32x4 raw = *(const u32x4*)&V[((size_t)(b * SEQ + kb * 64 + key)) * (NKV * HD) +
                                    kvh * HD + c];
      const bf16_t* e = (const bf16_t*)&raw;
#pragma unroll
      for (int j = 0; j < 8; ++j) Vt[c + j][key] = e[j];
    }
    __syncthreads();

    // scores S = Q K^T for 4 chunks of 16 keys
    v8f sc[4];
#pragma unroll
    for (int kc = 0; kc < 4; ++kc) {
      v8f a = {};
#pragma unroll
      for (int c = 0; c < 4; ++c) {
        Frag kf2;
        int key = kc * 16 + ln15;
        kf2.q[0] = *(const u32x4*)&Ks[key][c * 32 + hi * 8];
        kf2.q[1] = *(const u32x4*)&Ks[key][c * 32 + 16 + hi * 8];
        a = __builtin_amdgcn_wmma_f32_16x16x32_bf16(false, qfr[c].v, false, kf2.v,
                                                    (short)0, a, false, false);
      }
      sc[kc] = a;
    }

    // scale + causal mask
#pragma unroll
    for (int kc = 0; kc < 4; ++kc) {
      int key = kb * 64 + kc * 16 + ln15;
#pragma unroll
      for (int r = 0; r < 8; ++r) {
        int qrow = qbase + r + hi * 8;
        float sv = sc[kc][r] * ATTN_SCALE;
        sc[kc][r] = (key <= qrow) ? sv : -1e30f;
      }
    }

    // online softmax (row reductions across the 16-lane group)
#pragma unroll
    for (int r = 0; r < 8; ++r) {
      float bm = fmaxf(fmaxf(sc[0][r], sc[1][r]), fmaxf(sc[2][r], sc[3][r]));
#pragma unroll
      for (int m = 8; m >= 1; m >>= 1) bm = fmaxf(bm, __shfl_xor(bm, m, 32));
      float nm = fmaxf(m_run[r], bm);
      float alpha = __expf(m_run[r] - nm);
      m_run[r] = nm;
      float rs = 0.f;
#pragma unroll
      for (int kc = 0; kc < 4; ++kc) {
        float p = __expf(sc[kc][r] - nm);
        sc[kc][r] = p;
        rs += p;
      }
#pragma unroll
      for (int m = 8; m >= 1; m >>= 1) rs += __shfl_xor(rs, m, 32);
      l_run[r] = l_run[r] * alpha + rs;
#pragma unroll
      for (int nd = 0; nd < 8; ++nd) oacc[nd][r] *= alpha;
    }

    // P (C layout) -> per-wave LDS (row-major)
#pragma unroll
    for (int kc = 0; kc < 4; ++kc)
#pragma unroll
      for (int r = 0; r < 8; ++r)
        Ps[wave * 16 + r + hi * 8][kc * 16 + ln15] = f2bf(sc[kc][r]);

    asm volatile("s_wait_dscnt 0" ::: "memory");

    // O += P V
#pragma unroll
    for (int ch = 0; ch < 2; ++ch) {
      Frag pf;
      int prow = wave * 16 + ln15;
      pf.q[0] = *(const u32x4*)&Ps[prow][ch * 32 + hi * 8];
      pf.q[1] = *(const u32x4*)&Ps[prow][ch * 32 + 16 + hi * 8];
#pragma unroll
      for (int nd = 0; nd < 8; ++nd) {
        Frag vf2;
        int drow = nd * 16 + ln15;
        vf2.q[0] = *(const u32x4*)&Vt[drow][ch * 32 + hi * 8];
        vf2.q[1] = *(const u32x4*)&Vt[drow][ch * 32 + 16 + hi * 8];
        oacc[nd] = __builtin_amdgcn_wmma_f32_16x16x32_bf16(false, pf.v, false, vf2.v,
                                                           (short)0, oacc[nd], false, false);
      }
    }
    __syncthreads();
  }

  // normalize + store (B, S, NH*HD) bf16
#pragma unroll
  for (int nd = 0; nd < 8; ++nd)
#pragma unroll
    for (int r = 0; r < 8; ++r) {
      int qrow = qbase + r + hi * 8;
      int col  = h * HD + nd * 16 + ln15;
      O[((size_t)(b * SEQ) + qrow) * (NH * HD) + col] = f2bf(oacc[nd][r] / l_run[r]);
    }
}

// ---------------------------------------------------------------------------
// Host orchestration
// ---------------------------------------------------------------------------
extern "C" void kernel_launch(void* const* d_in, const int* in_sizes, int n_in,
                              void* d_out, int out_size, void* d_ws, size_t ws_size,
                              hipStream_t stream) {
  const float* x   = (const float*)d_in[0];
  // d_in[1] = attention_mask (pure causal; recomputed analytically)
  const float* wq  = (const float*)d_in[2];
  const float* wk  = (const float*)d_in[3];
  const float* wv  = (const float*)d_in[4];
  const float* wo  = (const float*)d_in[5];
  const float* qnw = (const float*)d_in[6];
  const float* knw = (const float*)d_in[7];
  const float* ln1 = (const float*)d_in[8];
  const float* ln2 = (const float*)d_in[9];
  const float* wg  = (const float*)d_in[10];
  const float* wu  = (const float*)d_in[11];
  const float* wd  = (const float*)d_in[12];
  float* out = (float*)d_out;

  char* p = (char*)d_ws;
  auto alloc = [&](size_t bytes) -> char* {
    char* r = p;
    p += (bytes + 255) & ~(size_t)255;
    return r;
  };

  const size_t NQD = (size_t)NH * HD;    // 4096
  const size_t NKD = (size_t)NKV * HD;   // 512

  bf16_t* wq_b  = (bf16_t*)alloc(NQD * HIDDEN * 2);
  bf16_t* wk_b  = (bf16_t*)alloc(NKD * HIDDEN * 2);
  bf16_t* wv_b  = (bf16_t*)alloc(NKD * HIDDEN * 2);
  bf16_t* wo_b  = (bf16_t*)alloc((size_t)HIDDEN * NQD * 2);
  bf16_t* wg_b  = (bf16_t*)alloc((size_t)DFF * HIDDEN * 2);
  bf16_t* wu_b  = (bf16_t*)alloc((size_t)DFF * HIDDEN * 2);
  bf16_t* wd_b  = (bf16_t*)alloc((size_t)HIDDEN * DFF * 2);
  bf16_t* xn_b  = (bf16_t*)alloc((size_t)MROWS * HIDDEN * 2);
  float*  q_f   = (float*)alloc((size_t)MROWS * NQD * 4);     // reused as act_b
  float*  k_f   = (float*)alloc((size_t)MROWS * NKD * 4);
  bf16_t* q_b   = (bf16_t*)alloc((size_t)MROWS * NQD * 2);
  bf16_t* k_b   = (bf16_t*)alloc((size_t)MROWS * NKD * 2);
  bf16_t* v_b   = (bf16_t*)alloc((size_t)MROWS * NKD * 2);
  bf16_t* at_b  = (bf16_t*)alloc((size_t)MROWS * NQD * 2);
  float*  h_f   = (float*)alloc((size_t)MROWS * HIDDEN * 4);
  bf16_t* hn_b  = xn_b;                  // reuse (xn consumed by QKV GEMMs)
  bf16_t* act_b = (bf16_t*)q_f;          // reuse (q_f consumed by rope kernel)

  auto cvt = [&](const float* src, bf16_t* dst, size_t n) {
    cvt_kernel<<<dim3((unsigned)((n + 1023) / 1024)), 256, 0, stream>>>(src, dst, (int)n);
  };
  cvt(wq, wq_b, NQD * HIDDEN);
  cvt(wk, wk_b, NKD * HIDDEN);
  cvt(wv, wv_b, NKD * HIDDEN);
  cvt(wo, wo_b, (size_t)HIDDEN * NQD);
  cvt(wg, wg_b, (size_t)DFF * HIDDEN);
  cvt(wu, wu_b, (size_t)DFF * HIDDEN);
  cvt(wd, wd_b, (size_t)HIDDEN * DFF);

  // h_norm = rms_norm(x, ln1)
  rmsnorm_kernel<<<MROWS, 256, 0, stream>>>(x, ln1, xn_b);

  // Q/K (f32, need norm+rope), V (bf16 direct)
  gemm_kernel<0><<<dim3(NQD / 64, MROWS / 128), 256, 0, stream>>>(
      xn_b, wq_b, nullptr, nullptr, q_f, nullptr, MROWS, (int)NQD, HIDDEN);
  gemm_kernel<0><<<dim3(NKD / 64, MROWS / 128), 256, 0, stream>>>(
      xn_b, wk_b, nullptr, nullptr, k_f, nullptr, MROWS, (int)NKD, HIDDEN);
  gemm_kernel<1><<<dim3(NKD / 64, MROWS / 128), 256, 0, stream>>>(
      xn_b, wv_b, nullptr, nullptr, nullptr, v_b, MROWS, (int)NKD, HIDDEN);

  // per-head RMSNorm + mRoPE -> bf16 (B,H,S,D)/(B,Hkv,S,D)
  {
    int rows = BATCH * SEQ * (NH + NKV);
    qknorm_rope_kernel<<<(rows + 7) / 8, 256, 0, stream>>>(q_f, k_f, qnw, knw, q_b, k_b);
  }

  // flash attention
  flash_kernel<<<dim3(SEQ / 128, NH, BATCH), 256, 0, stream>>>(q_b, k_b, v_b, at_b);

  // h = x + attn @ wo^T
  gemm_kernel<2><<<dim3(HIDDEN / 64, MROWS / 128), 256, 0, stream>>>(
      at_b, wo_b, nullptr, x, h_f, nullptr, MROWS, HIDDEN, (int)NQD);

  // hn = rms_norm(h, ln2)
  rmsnorm_kernel<<<MROWS, 256, 0, stream>>>(h_f, ln2, hn_b);

  // act = silu(hn @ wg^T) * (hn @ wu^T)
  gemm_kernel<3><<<dim3(DFF / 64, MROWS / 128), 256, 0, stream>>>(
      hn_b, wg_b, wu_b, nullptr, nullptr, act_b, MROWS, DFF, HIDDEN);

  // out = h + act @ wd^T
  gemm_kernel<2><<<dim3(HIDDEN / 64, MROWS / 128), 256, 0, stream>>>(
      act_b, wd_b, nullptr, h_f, out, nullptr, MROWS, HIDDEN, DFF);
}
